// BinaryLayer_72456098283938
// MI455X (gfx1250) — compile-verified, matrix operand order checked by reading, not run on hardware
//
#include <hip/hip_runtime.h>

// ---------------------------------------------------------------------------
// out = sign( x[8192x2048] @ sign(W[2048x2048]) )   (all f32 in/out)
// CDNA5 (gfx1250, wave32): bf16 WMMA GEMM with f32 accumulation.
//   v_wmma_f32_16x16x32_bf16, D = A(16x32) * B(32x16) + C(16x16 f32)
// Tiling: block = 128(M) x 256(N), 8 waves, each wave owns 64x64 (4x4 WMMA
// accumulators) -> 16 WMMAs per 8 LDS fragments per k-step (512 B LDS/WMMA).
// ---------------------------------------------------------------------------

typedef __attribute__((ext_vector_type(16))) __bf16 v16bf;
typedef __attribute__((ext_vector_type(8)))  float  v8f;

#define TILE_M 128
#define TILE_N 256
#define TILE_K 64          // two WMMA k-steps (K=32 each) per LDS stage
#define LDS_STRIDE 72      // 64 + 8 ushort pad -> spreads LDS banks

// fp32 -> bf16 with round-to-nearest-even (bit trick; NaN irrelevant here)
__device__ __forceinline__ unsigned short f32_to_bf16_rne(float f) {
    unsigned int u = __float_as_uint(f);
    u += 0x7fffu + ((u >> 16) & 1u);
    return (unsigned short)(u >> 16);
}

// sign(w) directly as a bf16 bit pattern: +1 = 0x3F80, -1 = 0xBF80, 0 = 0
__device__ __forceinline__ unsigned short sign_to_bf16(float f) {
    return (f > 0.0f) ? (unsigned short)0x3F80
         : (f < 0.0f) ? (unsigned short)0xBF80
                      : (unsigned short)0x0000;
}

struct Q2 { uint4 a; uint4 b; };   // 32 bytes == one v16bf fragment

// Per-lane WMMA 16-bit operand fragment gather from LDS (row-major, K contig).
// ISA layout (16-bit A 16x32): lanes 0-15 hold K={0..7,16..23} of row (lane&15),
// lanes 16-31 hold K={8..15,24..31}. Two contiguous 16B runs -> 2x ds_load_b128.
// B is stored transposed in LDS (ldsB[n][k]) so the same gather applies.
__device__ __forceinline__ v16bf load_frag(const unsigned short* p) {
    Q2 q;
    q.a = *(const uint4*)(p);        // K = base+0 .. base+7
    q.b = *(const uint4*)(p + 16);   // K = base+16 .. base+23
    return __builtin_bit_cast(v16bf, q);
}

__global__ __launch_bounds__(256)
void binary_gemm_sign_kernel(const float* __restrict__ x,
                             const float* __restrict__ w,
                             float* __restrict__ out,
                             int M, int N, int K) {
    __shared__ unsigned short ldsA[TILE_M * LDS_STRIDE]; // A tile, bf16, row-major [m][k]
    __shared__ unsigned short ldsB[TILE_N * LDS_STRIDE]; // B tile, bf16, TRANSPOSED [n][k]

    const int tid  = threadIdx.x;
    const int lane = tid & 31;
    const int wave = tid >> 5;          // 8 waves per block (2 x 4)
    const int half = lane >> 4;         // K-half select (ISA 16-bit operand layout)
    const int lrow = lane & 15;         // fragment row (A: m, B: n, C: n)

    const int wm = (wave & 1) * 64;     // wave's M offset inside 128x256 tile
    const int wn = (wave >> 1) * 64;    // wave's N offset inside 128x256 tile

    const int blockM = blockIdx.y * TILE_M;
    const int blockN = blockIdx.x * TILE_N;

    v8f acc[4][4];
    {
        v8f z = {};
#pragma unroll
        for (int i = 0; i < 4; ++i)
#pragma unroll
            for (int j = 0; j < 4; ++j) acc[i][j] = z;
    }

    for (int k0 = 0; k0 < K; k0 += TILE_K) {
        // ---- Stage A tile: 128 x 64 f32 -> bf16 LDS (2048 float4, 8/thread)
#pragma unroll
        for (int i = 0; i < 8; ++i) {
            int idx = tid + i * 256;                  // 0..2047
            int row = idx >> 4;                       // 16 float4 per row
            int c4  = (idx & 15) << 2;                // column in floats
            const float4 f = *(const float4*)(x + (size_t)(blockM + row) * K + k0 + c4);
            unsigned int lo = (unsigned int)f32_to_bf16_rne(f.x)
                            | ((unsigned int)f32_to_bf16_rne(f.y) << 16);
            unsigned int hi = (unsigned int)f32_to_bf16_rne(f.z)
                            | ((unsigned int)f32_to_bf16_rne(f.w) << 16);
            uint2 packed; packed.x = lo; packed.y = hi;
            *(uint2*)(&ldsA[row * LDS_STRIDE + c4]) = packed;  // 8B-aligned ds_store_b64
        }

        // ---- Stage B tile: 64 x 256 f32, binarize + transpose into ldsB[n][k]
#pragma unroll
        for (int i = 0; i < 16; ++i) {
            int idx = tid + i * 256;                  // 0..4095
            int kr  = idx >> 6;                       // 64 float4 per row of W
            int c4  = (idx & 63) << 2;                // column (n) in floats
            const float4 f = *(const float4*)(w + (size_t)(k0 + kr) * N + blockN + c4);
            ldsB[(c4 + 0) * LDS_STRIDE + kr] = sign_to_bf16(f.x);
            ldsB[(c4 + 1) * LDS_STRIDE + kr] = sign_to_bf16(f.y);
            ldsB[(c4 + 2) * LDS_STRIDE + kr] = sign_to_bf16(f.z);
            ldsB[(c4 + 3) * LDS_STRIDE + kr] = sign_to_bf16(f.w);
        }

        __syncthreads();

        // ---- Two WMMA k-steps over the staged 64-wide K slab
#pragma unroll
        for (int kk = 0; kk < TILE_K; kk += 32) {
            v16bf afrag[4];
            v16bf bfrag[4];
#pragma unroll
            for (int i = 0; i < 4; ++i)
                afrag[i] = load_frag(&ldsA[(wm + i * 16 + lrow) * LDS_STRIDE + kk + half * 8]);
#pragma unroll
            for (int j = 0; j < 4; ++j)
                bfrag[j] = load_frag(&ldsB[(wn + j * 16 + lrow) * LDS_STRIDE + kk + half * 8]);

#pragma unroll
            for (int i = 0; i < 4; ++i)
#pragma unroll
                for (int j = 0; j < 4; ++j)
                    acc[i][j] = __builtin_amdgcn_wmma_f32_16x16x32_bf16(
                        /*neg_a=*/false, afrag[i],
                        /*neg_b=*/false, bfrag[j],
                        /*c_mod=*/(short)0, acc[i][j],
                        /*reuse_a=*/false, /*reuse_b=*/false);
        }

        __syncthreads();
    }

    // ---- Epilogue: out = sign(acc)  (tf.sign semantics: sign(0) == 0)
    // C layout: VGPR r holds (M = r + 8*half, N = lane&15) per the ISA table.
#pragma unroll
    for (int i = 0; i < 4; ++i) {
#pragma unroll
        for (int j = 0; j < 4; ++j) {
            const int col = blockN + wn + j * 16 + lrow;
#pragma unroll
            for (int r = 0; r < 8; ++r) {
                const int row = blockM + wm + i * 16 + half * 8 + r;
                const float v = acc[i][j][r];
                out[(size_t)row * N + col] =
                    (v > 0.0f) ? 1.0f : ((v < 0.0f) ? -1.0f : 0.0f);
            }
        }
    }
}

extern "C" void kernel_launch(void* const* d_in, const int* in_sizes, int n_in,
                              void* d_out, int out_size, void* d_ws, size_t ws_size,
                              hipStream_t stream) {
    const float* x = (const float*)d_in[0];   // [M, K] f32
    const float* w = (const float*)d_in[1];   // [K, N] f32
    float* out     = (float*)d_out;           // [M, N] f32 in {-1,0,+1}

    const int K = 2048;
    const int N = 2048;
    const int M = in_sizes[0] / K;            // 8192 for the reference setup

    dim3 grid(N / TILE_N, M / TILE_M);        // (8, 64)
    dim3 block(256);                          // 8 wave32 waves
    binary_gemm_sign_kernel<<<grid, block, 0, stream>>>(x, w, out, M, N, K);
}
